// NounModule_53764400611393
// MI455X (gfx1250) — matrix operand order repaired and checked by reading, not compiled, not run on hardware
//
#include <hip/hip_runtime.h>
#include <hip/hip_bf16.h>
#include <stdint.h>

#define NROWS  32768   // B*S
#define DIN    1024
#define DCODE  256
#define KCODES 4096

typedef __attribute__((ext_vector_type(16))) __bf16 v16bf;
typedef __attribute__((ext_vector_type(8)))  float  v8f;

// ---- float -> bf16 round-to-nearest-even (manual, storage-type safe) ----
__device__ __forceinline__ __bf16 f2bf(float f) {
    union { float f; uint32_t u; } v; v.f = f;
    uint32_t r = (v.u + 0x7FFFu + ((v.u >> 16) & 1u)) >> 16;
    union { uint16_t s; __bf16 b; } o; o.s = (uint16_t)r;
    return o.b;
}

// ---- A fragment (16x32 bf16) from f32 row-major source ----
// lane L: row m = m0 + L%16, half = L/16
// vgpr v holds K = kt + (v&3)*2 + (v>>2)*16 + half*8  (pairs)
__device__ __forceinline__ v16bf load_a_f32(const float* __restrict__ A, int lda,
                                            int m0, int kt, int lane) {
    const int m = m0 + (lane & 15);
    const int h = lane >> 4;
    const float* row = A + (size_t)m * lda + kt + h * 8;
    v16bf a;
#pragma unroll
    for (int v = 0; v < 8; ++v) {
        const int k = (v & 3) * 2 + (v >> 2) * 16;
        float2 f = *(const float2*)(row + k);
        a[2 * v]     = f2bf(f.x);
        a[2 * v + 1] = f2bf(f.y);
    }
    return a;
}

// ---- A fragment (16x32 bf16) from a per-lane bf16 row pointer ----
__device__ __forceinline__ v16bf load_a_bf16_row(const __bf16* __restrict__ row,
                                                 int kt, int lane) {
    const int h = lane >> 4;
    const __bf16* p = row + kt + h * 8;
    v16bf a;
#pragma unroll
    for (int v = 0; v < 8; ++v) {
        const int k = (v & 3) * 2 + (v >> 2) * 16;
        a[2 * v]     = p[k];
        a[2 * v + 1] = p[k + 1];
    }
    return a;
}

// ---- B fragment (32x16 bf16) from B^T stored row-major [Ncols x Kdim] ----
// lane L: col n = n0 + L%16, half = L/16; element e<8 = Bt[n][kt + half*8 + e],
// e>=8 = Bt[n][kt + 16 + half*8 + (e-8)]  -> two 8-bf16 runs (16B aligned)
__device__ __forceinline__ v16bf load_bt(const __bf16* __restrict__ Bt, int ldk,
                                         int n0, int kt, int lane) {
    const int n = n0 + (lane & 15);
    const int h = lane >> 4;
    const __bf16* p = Bt + (size_t)n * ldk + kt + h * 8;
    v16bf b;
#pragma unroll
    for (int e = 0; e < 8; ++e) {
        b[e]     = p[e];
        b[e + 8] = p[16 + e];
    }
    return b;
}

__device__ __forceinline__ v8f wmma_bf16(v16bf a, v16bf b, v8f c) {
    return __builtin_amdgcn_wmma_f32_16x16x32_bf16(
        /*neg_a=*/false, a, /*neg_b=*/false, b,
        /*c_mod=*/(short)0, c, /*reuse_a=*/false, /*reuse_b=*/false);
}

// =================== prep kernels ===================

// w_to [DIN x DCODE] f32 -> w_toT [DCODE x DIN] bf16
__global__ __launch_bounds__(256) void k_prep_wto(const float* __restrict__ w,
                                                  __bf16* __restrict__ wT) {
    const int o = blockIdx.x * 256 + threadIdx.x;   // o = n*DIN + k
    const int n = o >> 10;
    const int k = o & 1023;
    wT[o] = f2bf(w[(size_t)k * DCODE + n]);
}

// w_from [DCODE x DIN] f32 -> w_fromT [DIN x DCODE] bf16
__global__ __launch_bounds__(256) void k_prep_wfrom(const float* __restrict__ w,
                                                    __bf16* __restrict__ wT) {
    const int o = blockIdx.x * 256 + threadIdx.x;   // o = n*DCODE + k
    const int n = o >> 8;
    const int k = o & 255;
    wT[o] = f2bf(w[(size_t)k * DIN + n]);
}

// codebook f32 -> bf16, plus half squared norms 0.5*||e_k||^2
__global__ __launch_bounds__(256) void k_prep_cb(const float* __restrict__ cb,
                                                 __bf16* __restrict__ cb_bf,
                                                 float* __restrict__ hn) {
    __shared__ float red[8];
    const int r = blockIdx.x;
    const int t = threadIdx.x;
    const float x = cb[(size_t)r * DCODE + t];
    cb_bf[(size_t)r * DCODE + t] = f2bf(x);
    float s = x * x;
#pragma unroll
    for (int off = 16; off > 0; off >>= 1) s += __shfl_xor(s, off, 32);
    if ((t & 31) == 0) red[t >> 5] = s;
    __syncthreads();
    if (t == 0) {
        float tot = 0.f;
#pragma unroll
        for (int i = 0; i < 8; ++i) tot += red[i];
        hn[r] = 0.5f * tot;
    }
}

// =================== encode GEMM: code = X @ w_to + b_to ===================
// wave tile 16x64, K = 1024 in 32-steps. 8 waves/block -> block tile 32x256.
__global__ __launch_bounds__(256) void k_encode(const float* __restrict__ X,
                                                const __bf16* __restrict__ WtoT,
                                                const float* __restrict__ b_to,
                                                __bf16* __restrict__ code) {
    const int lane = threadIdx.x & 31;
    const int wave = threadIdx.x >> 5;
    const int gw = blockIdx.x * 8 + wave;
    const int m0 = (gw >> 2) * 16;
    const int n0 = (gw & 3) * 64;

    v8f acc[4] = {};
    for (int kt = 0; kt < DIN; kt += 32) {
        v16bf a = load_a_f32(X, DIN, m0, kt, lane);
#pragma unroll
        for (int j = 0; j < 4; ++j) {
            v16bf b = load_bt(WtoT, DIN, n0 + j * 16, kt, lane);
            acc[j] = wmma_bf16(a, b, acc[j]);
        }
    }
    const int h = lane >> 4;
    const int nl = lane & 15;
#pragma unroll
    for (int j = 0; j < 4; ++j) {
        const int n = n0 + j * 16 + nl;
        const float bias = b_to[n];
#pragma unroll
        for (int v = 0; v < 8; ++v) {
            const int m = m0 + h * 8 + v;
            code[(size_t)m * DCODE + n] = f2bf(acc[j][v] + bias);
        }
    }
}

// ========== fused score GEMM + argmin: idx[m] = argmin_k (hn[k] - code.e_k) ==========
// wave owns 16 rows; preloads all 8 A-fragments (K=256) once, streams all 4096 codes.
__global__ __launch_bounds__(256) void k_score_argmin(const __bf16* __restrict__ code,
                                                      const __bf16* __restrict__ cb,
                                                      const float* __restrict__ hn,
                                                      int* __restrict__ idx) {
    const int lane = threadIdx.x & 31;
    const int wave = threadIdx.x >> 5;
    const int m0 = (blockIdx.x * 8 + wave) * 16;
    const int nl = lane & 15;

    const __bf16* arow = code + (size_t)(m0 + nl) * DCODE;
    v16bf afrag[8];
#pragma unroll
    for (int t = 0; t < 8; ++t) afrag[t] = load_a_bf16_row(arow, t * 32, lane);

    float best[8];
    int bidx[8];
#pragma unroll
    for (int v = 0; v < 8; ++v) { best[v] = 3.4e38f; bidx[v] = 0; }

    for (int n0 = 0; n0 < KCODES; n0 += 16) {
        v8f c = {};
#pragma unroll
        for (int t = 0; t < 8; ++t) {
            v16bf b = load_bt(cb, DCODE, n0, t * 32, lane);
            c = wmma_bf16(afrag[t], b, c);
        }
        const int n = n0 + nl;
        const float h2 = hn[n];
#pragma unroll
        for (int v = 0; v < 8; ++v) {
            const float d = h2 - c[v];
            const bool take = d < best[v];         // strict <: keeps first index
            best[v] = take ? d : best[v];
            bidx[v] = take ? n : bidx[v];
        }
    }
    // reduce across the 16 lanes of each half (offsets < 16 never cross halves)
#pragma unroll
    for (int v = 0; v < 8; ++v) {
#pragma unroll
        for (int off = 8; off > 0; off >>= 1) {
            const float ob = __shfl_xor(best[v], off, 32);
            const int   oi = __shfl_xor(bidx[v], off, 32);
            const bool take = (ob < best[v]) || (ob == best[v] && oi < bidx[v]);
            best[v] = take ? ob : best[v];
            bidx[v] = take ? oi : bidx[v];
        }
    }
    if (nl == 0) {
        const int h = lane >> 4;
#pragma unroll
        for (int v = 0; v < 8; ++v) idx[m0 + h * 8 + v] = bidx[v];
    }
}

// ========== decode GEMM: out = codebook[idx] @ w_from + b_from ==========
// wave tile 16x64, K = 256 in 8 steps; A gathered by per-row codebook pointer.
__global__ __launch_bounds__(256) void k_decode(const int* __restrict__ idx,
                                                const __bf16* __restrict__ cb,
                                                const __bf16* __restrict__ WfromT,
                                                const float* __restrict__ b_from,
                                                float* __restrict__ out) {
    const int lane = threadIdx.x & 31;
    const int wave = threadIdx.x >> 5;
    const int gw = blockIdx.x * 8 + wave;
    const int m0 = (gw >> 4) * 16;
    const int n0 = (gw & 15) * 64;
    const int nl = lane & 15;

    const __bf16* arow = cb + (size_t)idx[m0 + nl] * DCODE;

    v8f acc[4] = {};
#pragma unroll
    for (int t = 0; t < 8; ++t) {
        v16bf a = load_a_bf16_row(arow, t * 32, lane);
#pragma unroll
        for (int j = 0; j < 4; ++j) {
            v16bf b = load_bt(WfromT, DCODE, n0 + j * 16, t * 32, lane);
            acc[j] = wmma_bf16(a, b, acc[j]);
        }
    }
    const int h = lane >> 4;
#pragma unroll
    for (int j = 0; j < 4; ++j) {
        const int n = n0 + j * 16 + nl;
        const float bias = b_from[n];
#pragma unroll
        for (int v = 0; v < 8; ++v) {
            out[(size_t)(m0 + h * 8 + v) * DIN + n] = acc[j][v] + bias;
        }
    }
}

// second tuple output: indices, written after the [N x DIN] f32 block
__global__ __launch_bounds__(256) void k_write_idx(const int* __restrict__ idx,
                                                   float* __restrict__ outf) {
    const int i = blockIdx.x * 256 + threadIdx.x;
    outf[i] = (float)idx[i];
}

extern "C" void kernel_launch(void* const* d_in, const int* in_sizes, int n_in,
                              void* d_out, int out_size, void* d_ws, size_t ws_size,
                              hipStream_t stream) {
    const float* features = (const float*)d_in[0];  // [8,4096,1024]
    const float* codebook = (const float*)d_in[1];  // [4096,256]
    const float* w_to     = (const float*)d_in[2];  // [1024,256]
    const float* b_to     = (const float*)d_in[3];  // [256]
    const float* w_from   = (const float*)d_in[4];  // [256,1024]
    const float* b_from   = (const float*)d_in[5];  // [1024]
    float* out = (float*)d_out;

    // workspace layout (~20 MB)
    char* ws = (char*)d_ws;
    __bf16* code_bf = (__bf16*)ws; ws += (size_t)NROWS  * DCODE * 2;  // 16 MB
    __bf16* cb_bf   = (__bf16*)ws; ws += (size_t)KCODES * DCODE * 2;  //  2 MB
    __bf16* wtoT    = (__bf16*)ws; ws += (size_t)DCODE  * DIN   * 2;  // .5 MB
    __bf16* wfromT  = (__bf16*)ws; ws += (size_t)DIN    * DCODE * 2;  // .5 MB
    float*  cb_hn   = (float*)ws;  ws += (size_t)KCODES * 4;
    int*    idx     = (int*)ws;

    hipLaunchKernelGGL(k_prep_wto,   dim3(1024),   dim3(256), 0, stream, w_to, wtoT);
    hipLaunchKernelGGL(k_prep_wfrom, dim3(1024),   dim3(256), 0, stream, w_from, wfromT);
    hipLaunchKernelGGL(k_prep_cb,    dim3(KCODES), dim3(256), 0, stream, codebook, cb_bf, cb_hn);

    // encode: 8192 wave tiles (16x64) -> 1024 blocks
    hipLaunchKernelGGL(k_encode, dim3(1024), dim3(256), 0, stream,
                       features, wtoT, b_to, code_bf);
    // score+argmin: 2048 wave tiles (16 rows each) -> 256 blocks
    hipLaunchKernelGGL(k_score_argmin, dim3(256), dim3(256), 0, stream,
                       code_bf, cb_bf, cb_hn, idx);
    // decode: 32768 wave tiles (16x64) -> 4096 blocks
    hipLaunchKernelGGL(k_decode, dim3(4096), dim3(256), 0, stream,
                       idx, cb_bf, wfromT, b_from, out);
    // indices tail
    hipLaunchKernelGGL(k_write_idx, dim3(NROWS / 256), dim3(256), 0, stream,
                       idx, out + (size_t)NROWS * DIN);
}
